// Block_47175920779550
// MI455X (gfx1250) — compile-verified, matrix operand order checked by reading, not run on hardware
//
#include <hip/hip_runtime.h>
#include <math.h>

// ---------------- problem constants (RWKV block) ----------------
#define Bb     8
#define Tt     2048
#define Cdim   1024
#define DAdim  1024
#define DFdim  4096
#define BT     (Bb * Tt)          // 16384 rows

// ---------------- vector types for WMMA ----------------
typedef __attribute__((ext_vector_type(16))) __bf16 v16bf;
typedef __attribute__((ext_vector_type(8)))  __bf16 v8bf;
typedef __attribute__((ext_vector_type(8)))  float  v8f;

__device__ __forceinline__ __bf16 f2bf(float f) {
  union { float f; unsigned u; } x; x.f = f;
  unsigned r = (x.u + 0x7FFFu + ((x.u >> 16) & 1u)) >> 16;   // round-nearest-even
  unsigned short hs = (unsigned short)r;
  __bf16 b;
  __builtin_memcpy(&b, &hs, 2);
  return b;
}

__device__ __forceinline__ void wait_async0() {
#if __has_builtin(__builtin_amdgcn_s_wait_asynccnt)
  __builtin_amdgcn_s_wait_asynccnt(0);
#else
  asm volatile("s_wait_asynccnt 0x0" ::: "memory");
#endif
}

// async copy of 16 bytes global -> LDS (gfx1250, tracked by ASYNCcnt).
// lds_lo32: low 32 bits of the generic LDS pointer == wave-relative LDS byte addr.
__device__ __forceinline__ void async_g2l_b128(const void* gptr, void* lptr) {
  unsigned lds = (unsigned)(unsigned long long)lptr;
  unsigned long long ga = (unsigned long long)gptr;
  asm volatile("global_load_async_to_lds_b128 %0, %1, off"
               :: "v"(lds), "v"(ga) : "memory");
}

// ---------------- f32 -> bf16 weight conversion ----------------
__global__ __launch_bounds__(256)
void cvt_bf16(const float* __restrict__ in, __bf16* __restrict__ outp) {
  size_t i = (size_t)blockIdx.x * 256 + threadIdx.x;
  outp[i] = f2bf(in[i]);
}

// ---------------- LayerNorm over rows of length Cdim ----------------
__global__ __launch_bounds__(256)
void ln_rows(const float* __restrict__ x, const float* __restrict__ g,
             const float* __restrict__ b, float* __restrict__ h) {
  __shared__ float sh[256];
  const int tid = threadIdx.x;
  const size_t row = blockIdx.x;
  const float* xr = x + row * Cdim;
  float vals[4];
  float s = 0.f, s2 = 0.f;
#pragma unroll
  for (int i = 0; i < 4; ++i) {
    float vv = xr[tid + 256 * i];
    vals[i] = vv; s += vv; s2 += vv * vv;
  }
  sh[tid] = s; __syncthreads();
  for (int o = 128; o > 0; o >>= 1) { if (tid < o) sh[tid] += sh[tid + o]; __syncthreads(); }
  float mean = sh[0] * (1.f / Cdim);
  __syncthreads();
  sh[tid] = s2; __syncthreads();
  for (int o = 128; o > 0; o >>= 1) { if (tid < o) sh[tid] += sh[tid + o]; __syncthreads(); }
  float var  = sh[0] * (1.f / Cdim) - mean * mean;
  float rstd = rsqrtf(var + 1e-5f);
  float* hr = h + row * Cdim;
#pragma unroll
  for (int i = 0; i < 4; ++i) {
    int c = tid + 256 * i;
    hr[c] = (vals[i] - mean) * rstd * g[c] + b[c];
  }
}

// ---------------- time-shift mix + bf16 pack (up to 3 outputs) ----------------
__global__ __launch_bounds__(256)
void mix_pack(const float* __restrict__ h,
              const float* __restrict__ c0, const float* __restrict__ c1,
              const float* __restrict__ c2,
              __bf16* __restrict__ o0, __bf16* __restrict__ o1, __bf16* __restrict__ o2) {
  size_t idx = (size_t)blockIdx.x * 256 + threadIdx.x;   // over BT*Cdim
  int c = (int)(idx & (Cdim - 1));
  size_t row = idx >> 10;                                // / Cdim
  int t = (int)(row & (Tt - 1));
  float hv = h[idx];
  float hp = (t == 0) ? 0.f : h[idx - Cdim];
  float a0 = c0[c];
  o0[idx] = f2bf(hv * a0 + hp * (1.f - a0));
  if (c1) { float a1 = c1[c]; o1[idx] = f2bf(hv * a1 + hp * (1.f - a1)); }
  if (c2) { float a2 = c2[c]; o2[idx] = f2bf(hv * a2 + hp * (1.f - a2)); }
}

// ---------------- numerically-stable WKV scan + sigmoid(r) gate ----------------
__global__ __launch_bounds__(256)
void wkv_scan(const float* __restrict__ td, const float* __restrict__ tf,
              const float* __restrict__ k, const float* __restrict__ v,
              const float* __restrict__ r, __bf16* __restrict__ y) {
  int gid = blockIdx.x * 256 + threadIdx.x;     // 0 .. Bb*DAdim-1
  int d  = gid & (DAdim - 1);
  int bb = gid >> 10;                           // / DAdim
  float w_ = -__expf(td[d]);
  float u  = tf[d];
  float a = 0.f, bacc = 0.f, p = -1e30f;
  size_t base = (size_t)bb * Tt * DAdim + d;
  for (int t = 0; t < Tt; ++t) {
    size_t idx = base + (size_t)t * DAdim;
    float kt = k[idx], vt = v[idx], rt = r[idx];
    float ww = u + kt;
    float q  = fmaxf(p, ww);
    float e1 = __expf(p - q);
    float e2 = __expf(ww - q);
    float yv = (e1 * a + e2 * vt) / (e1 * bacc + e2);
    float sg = 1.f / (1.f + __expf(-rt));
    y[idx] = f2bf(yv * sg);
    float ww2 = p + w_;
    float q2  = fmaxf(ww2, kt);
    float e1b = __expf(ww2 - q2);
    float e2b = __expf(kt - q2);
    a    = e1b * a + e2b * vt;
    bacc = e1b * bacc + e2b;
    p    = q2;
  }
}

// ---------------- bf16 WMMA GEMM: out[m,n] = sum_k A[m,k] * W[n,k] ----------------
// Block: 256 threads (8 waves). Tile 128(M) x 128(N) x 64(K), LDS double-buffered,
// filled with global_load_async_to_lds_b128 (ASYNCcnt) so copies of tile t+1 overlap
// the 16 v_wmma_f32_16x16x32_bf16 of tile t. One workgroup barrier per K-tile.
// Wave grid 2(M) x 4(N); each wave owns a 64x32 output (4x2 WMMA accumulators).
// Epilogues: 0 = f32 store; 1 = res + acc (f32); 2 = relu(acc)^2 -> bf16;
//            3 = res + sigmoid(gate) * acc (f32, res may alias outF).
#define BLK_M 128
#define BLK_N 128
#define BLK_K 64
#define LDSK  72   // padded LDS row stride (bf16 elems) to break bank conflicts

template <int EPI>
__global__ __launch_bounds__(256)
void gemm_bf16(const __bf16* __restrict__ A, const __bf16* __restrict__ W,
               int Kdim, int Ndim,
               float* __restrict__ outF, __bf16* __restrict__ outB,
               const float* __restrict__ res, const float* __restrict__ gate) {
  __shared__ __bf16 As[2][BLK_M][LDSK];
  __shared__ __bf16 Bs[2][BLK_N][LDSK];

  const int tid  = threadIdx.x;
  const int lane = tid & 31;
  const int wave = tid >> 5;
  const int wm   = wave >> 2;        // 0..1
  const int wn   = wave & 3;         // 0..3
  const int r16  = lane & 15;
  const int half = lane >> 4;
  const size_t m0 = (size_t)blockIdx.y * BLK_M;
  const int    n0 = blockIdx.x * BLK_N;

  v8f acc[4][2];
#pragma unroll
  for (int i = 0; i < 4; ++i)
#pragma unroll
    for (int j = 0; j < 2; ++j)
#pragma unroll
      for (int e = 0; e < 8; ++e) acc[i][j][e] = 0.f;

  // Async-stage one 128x64 A tile + 128x64 W tile into LDS buffer `buf`.
  // 128 rows * 8 chunks(8 bf16) = 1024 chunks per array; 256 threads x 4.
  auto issue_tile = [&](int buf, int k0) {
#pragma unroll
    for (int i = 0; i < 4; ++i) {
      int q  = tid + 256 * i;
      int rr = q >> 3;
      int cc = (q & 7) << 3;
      async_g2l_b128(A + (m0 + rr) * (size_t)Kdim + k0 + cc, &As[buf][rr][cc]);
      async_g2l_b128(W + ((size_t)(n0 + rr)) * (size_t)Kdim + k0 + cc, &Bs[buf][rr][cc]);
    }
    // speculative prefetch of the tile after next into GL2
    __builtin_prefetch(A + (m0 + (tid >> 2)) * (size_t)Kdim + k0 + BLK_K, 0, 1);
    __builtin_prefetch(W + ((size_t)(n0 + (tid >> 2))) * (size_t)Kdim + k0 + BLK_K, 0, 1);
  };

  // Fragments per ISA 16-bit A/B layout: lane holds row r16, K-halves selected by
  // (lane>>4): two contiguous 16B chunks at K = kk+8*half and kk+16+8*half.
  auto compute_tile = [&](int buf) {
#pragma unroll
    for (int kk = 0; kk < BLK_K; kk += 32) {
      v16bf af[4], bfrag[2];
#pragma unroll
      for (int i = 0; i < 4; ++i) {
        const __bf16* p = &As[buf][wm * 64 + i * 16 + r16][kk + half * 8];
        v8bf lo = *(const v8bf*)p;
        v8bf hi = *(const v8bf*)(p + 16);
        af[i] = __builtin_shufflevector(lo, hi, 0, 1, 2, 3, 4, 5, 6, 7,
                                                8, 9, 10, 11, 12, 13, 14, 15);
      }
#pragma unroll
      for (int j = 0; j < 2; ++j) {
        const __bf16* p = &Bs[buf][wn * 32 + j * 16 + r16][kk + half * 8];
        v8bf lo = *(const v8bf*)p;
        v8bf hi = *(const v8bf*)(p + 16);
        bfrag[j] = __builtin_shufflevector(lo, hi, 0, 1, 2, 3, 4, 5, 6, 7,
                                                   8, 9, 10, 11, 12, 13, 14, 15);
      }
#pragma unroll
      for (int i = 0; i < 4; ++i)
#pragma unroll
        for (int j = 0; j < 2; ++j)
          acc[i][j] = __builtin_amdgcn_wmma_f32_16x16x32_bf16(
              false, af[i], false, bfrag[j], (short)0, acc[i][j], false, false);
    }
  };

  // Software pipeline: K/BLK_K is 16 or 64 here (always even), so unroll in
  // buffer pairs to keep the LDS buffer index a compile-time constant.
  issue_tile(0, 0);
  for (int k0 = 0; k0 < Kdim; k0 += 2 * BLK_K) {
    wait_async0();
    __syncthreads();
    if (k0 + BLK_K < Kdim) issue_tile(1, k0 + BLK_K);
    compute_tile(0);
    wait_async0();
    __syncthreads();
    if (k0 + 2 * BLK_K < Kdim) issue_tile(0, k0 + 2 * BLK_K);
    compute_tile(1);
  }

  // Epilogue. C/D layout: VGPR e -> row = 16*sub_m + e + 8*half; col = 16*sub_n + r16.
#pragma unroll
  for (int i = 0; i < 4; ++i) {
#pragma unroll
    for (int j = 0; j < 2; ++j) {
      size_t mbase = m0 + wm * 64 + i * 16 + half * 8;
      int n = n0 + wn * 32 + j * 16 + r16;
#pragma unroll
      for (int e = 0; e < 8; ++e) {
        size_t idx = (mbase + e) * (size_t)Ndim + n;
        float v = acc[i][j][e];
        if (EPI == 0) {
          outF[idx] = v;
        } else if (EPI == 1) {
          outF[idx] = res[idx] + v;
        } else if (EPI == 2) {
          float t = v > 0.f ? v : 0.f;
          outB[idx] = f2bf(t * t);
        } else {
          float s = 1.f / (1.f + __expf(-gate[idx]));
          outF[idx] = res[idx] + s * v;
        }
      }
    }
  }
}

// ---------------- host-side orchestration ----------------
extern "C" void kernel_launch(void* const* d_in, const int* in_sizes, int n_in,
                              void* d_out, int out_size, void* d_ws, size_t ws_size,
                              hipStream_t stream) {
  (void)in_sizes; (void)n_in; (void)out_size; (void)ws_size;
  const float* x          = (const float*)d_in[0];
  const float* ln1_g      = (const float*)d_in[1];
  const float* ln1_b      = (const float*)d_in[2];
  const float* ln2_g      = (const float*)d_in[3];
  const float* ln2_b      = (const float*)d_in[4];
  const float* tm_k       = (const float*)d_in[5];
  const float* tm_v       = (const float*)d_in[6];
  const float* tm_r       = (const float*)d_in[7];
  const float* time_decay = (const float*)d_in[8];
  const float* time_first = (const float*)d_in[9];
  const float* Wk         = (const float*)d_in[10];
  const float* Wv         = (const float*)d_in[11];
  const float* Wr         = (const float*)d_in[12];
  const float* Wo         = (const float*)d_in[13];
  const float* fm_k       = (const float*)d_in[14];
  const float* fm_r       = (const float*)d_in[15];
  const float* Fk         = (const float*)d_in[16];
  const float* Fr         = (const float*)d_in[17];
  const float* Fv         = (const float*)d_in[18];
  float* out = (float*)d_out;

  char* wsp = (char*)d_ws;
  size_t off = 0;
  auto alloc = [&](size_t bytes) -> char* {
    char* p = wsp + off;
    off += (bytes + 255) & ~(size_t)255;
    return p;
  };

  __bf16* Wk_b = (__bf16*)alloc((size_t)DAdim * Cdim  * 2);
  __bf16* Wv_b = (__bf16*)alloc((size_t)DAdim * Cdim  * 2);
  __bf16* Wr_b = (__bf16*)alloc((size_t)DAdim * Cdim  * 2);
  __bf16* Wo_b = (__bf16*)alloc((size_t)Cdim  * DAdim * 2);
  __bf16* Fk_b = (__bf16*)alloc((size_t)DFdim * Cdim  * 2);
  __bf16* Fr_b = (__bf16*)alloc((size_t)Cdim  * Cdim  * 2);
  __bf16* Fv_b = (__bf16*)alloc((size_t)Cdim  * DFdim * 2);
  float*  hbuf = (float*) alloc((size_t)BT * Cdim  * 4);
  __bf16* xk   = (__bf16*)alloc((size_t)BT * Cdim  * 2);
  __bf16* xv   = (__bf16*)alloc((size_t)BT * Cdim  * 2);
  __bf16* xr   = (__bf16*)alloc((size_t)BT * Cdim  * 2);
  float*  kbuf = (float*) alloc((size_t)BT * DAdim * 4);
  float*  vbuf = (float*) alloc((size_t)BT * DAdim * 4);
  float*  rbuf = (float*) alloc((size_t)BT * DAdim * 4);
  __bf16* rwkv = (__bf16*)alloc((size_t)BT * DAdim * 2);
  __bf16* kf   = (__bf16*)alloc((size_t)BT * DFdim * 2);
  float*  rr   = kbuf;   // reuse: kbuf is dead after the WKV scan

  // 1) weights -> bf16
  cvt_bf16<<<(DAdim * Cdim)  / 256, 256, 0, stream>>>(Wk, Wk_b);
  cvt_bf16<<<(DAdim * Cdim)  / 256, 256, 0, stream>>>(Wv, Wv_b);
  cvt_bf16<<<(DAdim * Cdim)  / 256, 256, 0, stream>>>(Wr, Wr_b);
  cvt_bf16<<<(Cdim * DAdim)  / 256, 256, 0, stream>>>(Wo, Wo_b);
  cvt_bf16<<<(DFdim * Cdim)  / 256, 256, 0, stream>>>(Fk, Fk_b);
  cvt_bf16<<<(Cdim * Cdim)   / 256, 256, 0, stream>>>(Fr, Fr_b);
  cvt_bf16<<<(Cdim * DFdim)  / 256, 256, 0, stream>>>(Fv, Fv_b);

  // 2) TimeMix front: LN1 + shift-mix
  ln_rows<<<BT, 256, 0, stream>>>(x, ln1_g, ln1_b, hbuf);
  mix_pack<<<(BT * Cdim) / 256, 256, 0, stream>>>(hbuf, tm_k, tm_v, tm_r, xk, xv, xr);

  // 3) k / v / r projections (WMMA)
  dim3 gKV(DAdim / BLK_N, BT / BLK_M);
  gemm_bf16<0><<<gKV, 256, 0, stream>>>(xk, Wk_b, Cdim, DAdim, kbuf, nullptr, nullptr, nullptr);
  gemm_bf16<0><<<gKV, 256, 0, stream>>>(xv, Wv_b, Cdim, DAdim, vbuf, nullptr, nullptr, nullptr);
  gemm_bf16<0><<<gKV, 256, 0, stream>>>(xr, Wr_b, Cdim, DAdim, rbuf, nullptr, nullptr, nullptr);

  // 4) WKV recurrence + sigmoid(r) gate
  wkv_scan<<<(Bb * DAdim) / 256, 256, 0, stream>>>(time_decay, time_first, kbuf, vbuf, rbuf, rwkv);

  // 5) output projection + residual -> d_out
  dim3 gWo(Cdim / BLK_N, BT / BLK_M);
  gemm_bf16<1><<<gWo, 256, 0, stream>>>(rwkv, Wo_b, DAdim, Cdim, out, nullptr, x, nullptr);

  // 6) ChannelMix front: LN2 + shift-mix (reuse hbuf / xk / xr)
  ln_rows<<<BT, 256, 0, stream>>>(out, ln2_g, ln2_b, hbuf);
  mix_pack<<<(BT * Cdim) / 256, 256, 0, stream>>>(hbuf, fm_k, fm_r, nullptr, xk, xr, nullptr);

  // 7) kf = relu(gk @ Fk^T)^2  (bf16 out)
  dim3 gFk(DFdim / BLK_N, BT / BLK_M);
  gemm_bf16<2><<<gFk, 256, 0, stream>>>(xk, Fk_b, Cdim, DFdim, nullptr, kf, nullptr, nullptr);

  // 8) rr = gr @ Fr^T (raw logits, f32)
  dim3 gFr(Cdim / BLK_N, BT / BLK_M);
  gemm_bf16<0><<<gFr, 256, 0, stream>>>(xr, Fr_b, Cdim, Cdim, rr, nullptr, nullptr, nullptr);

  // 9) d_out += sigmoid(rr) * (kf @ Fv^T)
  dim3 gFv(Cdim / BLK_N, BT / BLK_M);
  gemm_bf16<3><<<gFv, 256, 0, stream>>>(kf, Fv_b, DFdim, Cdim, out, nullptr, out, rr);
}
